// positional_VQVAE_63196148793711
// MI455X (gfx1250) — matrix operand order repaired and checked by reading, not compiled, not run on hardware
//
#include <hip/hip_runtime.h>
#include <hip/hip_bf16.h>
#include <stdint.h>

// ---------------------------------------------------------------------------
// CDNA5 (gfx1250) VQ-VAE forward.  All GEMM-like work (convs, attention,
// linears) goes through v_wmma_f32_16x16x32_bf16 (fp32 accumulate, bf16
// operands converted in registers; fp32 kept in HBM).  VQ codebook search
// stages the codebook in LDS via global_load_async_to_lds_b128.
// All gather guards are branch-free: 32-bit offsets selected with cndmask.
// Store epilogues use no division (M-tiles never cross a batch boundary).
// ---------------------------------------------------------------------------

typedef __attribute__((ext_vector_type(16))) __bf16 bf16x16;
typedef __attribute__((ext_vector_type(8)))  float  f32x8;

__device__ __forceinline__ f32x8 wmma_bf16(bf16x16 a, bf16x16 b, f32x8 c) {
  // D = A(16x32) * B(32x16) + C, fp32 accumulate
  return __builtin_amdgcn_wmma_f32_16x16x32_bf16(false, a, false, b, (short)0, c, false, false);
}

// A-fragment K index for element i of the 16-elem per-lane vector
// (ISA 7.12.2: lanes 0-15 hold K 0..7 & 16..23, lanes 16-31 hold 8..15 & 24..31)
__device__ __forceinline__ int akmap(int i, int half) {
  return (i < 8) ? (half * 8 + i) : (8 + half * 8 + i);
}
// B-fragment K index (same as A: lanes 0-15: K 0..7 & 16..23; lanes 16-31: K 8..15 & 24..31)
__device__ __forceinline__ int bkmap(int i, int half) { return (i < 8) ? (half * 8 + i) : (8 + half * 8 + i); }

// ---------------------------------------------------------------------------
// GroupNorm (32 groups) + optional SiLU.  One block per (batch, group).
// ---------------------------------------------------------------------------
__global__ void gn_kernel(const float* __restrict__ x, const float* __restrict__ gamma,
                          const float* __restrict__ beta, float* __restrict__ y,
                          int C, int HW, int dosilu)
{
  const int b   = blockIdx.x >> 5;
  const int grp = blockIdx.x & 31;
  const int cpg = C >> 5;
  const long base = ((long)b * C + (long)grp * cpg) * HW;
  const int n = cpg * HW;

  float s = 0.f, ss = 0.f;
  for (int i = threadIdx.x; i < n; i += 256) {
    float v = x[base + i];
    s += v; ss += v * v;
  }
  __shared__ float r1[256], r2[256];
  r1[threadIdx.x] = s; r2[threadIdx.x] = ss;
  __syncthreads();
  for (int st = 128; st > 0; st >>= 1) {
    if (threadIdx.x < st) { r1[threadIdx.x] += r1[threadIdx.x + st];
                            r2[threadIdx.x] += r2[threadIdx.x + st]; }
    __syncthreads();
  }
  const float mean = r1[0] / n;
  const float var  = r2[0] / n - mean * mean;
  const float rstd = rsqrtf(var + 1e-5f);
  for (int c = 0; c < cpg; ++c) {            // channel loop hoisted: no div in hot loop
    const float ga = gamma[grp * cpg + c];
    const float be = beta[grp * cpg + c];
    const long cb = base + (long)c * HW;
    for (int i = threadIdx.x; i < HW; i += 256) {
      float v = (x[cb + i] - mean) * rstd * ga + be;
      if (dosilu) v = v / (1.f + __expf(-v));
      y[cb + i] = v;
    }
  }
}

// ---------------------------------------------------------------------------
// Implicit-GEMM conv2d (NCHW, OIHW weights), bf16 WMMA, fp32 accumulate.
// Each wave: one 16(M=spatial)x16(N=Cout) tile; K = Cin*KH*KW in chunks of 32.
// HoWo is a multiple of 16 for every layer -> a tile stays in one batch and
// the store epilogue needs no integer division.
// Optional fused residual add.
// ---------------------------------------------------------------------------
template<int KH, int KW>
__global__ void __launch_bounds__(128)
conv2d_wmma(const float* __restrict__ in, const float* __restrict__ wt,
            const float* __restrict__ bias, const float* __restrict__ resid,
            float* __restrict__ out,
            int B, int Cin, int H, int Wd, int Cout, int Ho, int Wo,
            int stride, int pad)
{
  constexpr int KK = KH * KW;
  const int lane = threadIdx.x & 31;
  const int wave = threadIdx.x >> 5;
  const int half = lane >> 4;
  const int mr   = lane & 15;
  const int nc   = lane & 15;
  const int m0 = blockIdx.x * 16;
  const int n0 = (blockIdx.y * 4 + wave) * 16;
  if (n0 >= Cout) return;                     // wave-uniform exit (EXEC stays all-1)

  const int HoWo = Ho * Wo;
  const int bb  = m0 / HoWo;                  // uniform: whole tile in one batch
  const int rs0 = m0 - bb * HoWo;             // tile start within batch
  const int rs  = rs0 + mr;
  const int oh  = rs / Wo;
  const int ow  = rs - oh * Wo;
  const int oh0 = oh * stride - pad;
  const int ow0 = ow * stride - pad;
  const int abase = bb * Cin * H * Wd;        // fits in i32 (max ~33.5M)

  const int Ktot = Cin * KK;
  const int oc   = n0 + nc;
  const bool vn  = oc < Cout;
  const int wbase = (vn ? oc : Cout - 1) * Ktot;
  __builtin_prefetch(wt + wbase, 0, 1);       // global_prefetch_b8

  f32x8 acc = {};
  for (int kb = 0; kb < Ktot; kb += 32) {
    bf16x16 af, bf;
#pragma unroll
    for (int i = 0; i < 16; ++i) {
      // ---- A: gathered input patch element (branch-free: i32 cndmask select)
      int ka = kb + akmap(i, half);
      int ci = ka / KK;
      int rr = ka - ci * KK;
      int kh = rr / KW;
      int kw = rr - kh * KW;
      int ih = oh0 + kh, iw = ow0 + kw;
      bool ok = (ka < Ktot) & ((unsigned)ih < (unsigned)H) & ((unsigned)iw < (unsigned)Wd);
      int aoff = abase + (ci * H + ih) * Wd + iw;
      aoff = ok ? aoff : 0;                   // v_cndmask_b32
      float av = in[aoff];
      af[i] = (__bf16)(ok ? av : 0.f);
      // ---- B: weight (OIHW is linear in k for fixed oc)
      int kB = kb + bkmap(i, half);
      bool okb = (kB < Ktot) & vn;
      int boff = wbase + kB;
      boff = okb ? boff : 0;                  // v_cndmask_b32
      float bv = wt[boff];
      bf[i] = (__bf16)(okb ? bv : 0.f);
    }
    acc = wmma_bf16(af, bf, acc);
  }
  // Store: D row r -> m = m0 + r + half*8, same batch for all rows.
  const long obase = ((long)bb * Cout + oc) * (long)HoWo + rs0 + half * 8;
  if (vn) {
    const float bsv = bias[oc];
#pragma unroll
    for (int r = 0; r < 8; ++r) {
      long idx = obase + r;
      float v = acc[r] + bsv;
      if (resid) v += resid[idx];
      out[idx] = v;
    }
  }
}

// ---------------------------------------------------------------------------
// ConvTranspose2d k=4 s=2 p=1 (torch weight layout [Cin,Cout,4,4]) via WMMA.
// out(oh,ow) gathers x(ih,iw) with ih=(oh+1-kh)/2 when even & in range.
// Cin/Cout are 256 here, so K (=Cin*16) and N need no remainder guards.
// ---------------------------------------------------------------------------
__global__ void __launch_bounds__(128)
convt_wmma(const float* __restrict__ in, const float* __restrict__ wt,
           const float* __restrict__ bias, float* __restrict__ out,
           int B, int Cin, int H, int Wd, int Cout)
{
  const int lane = threadIdx.x & 31;
  const int wave = threadIdx.x >> 5;
  const int half = lane >> 4;
  const int mr   = lane & 15;
  const int nc   = lane & 15;
  const int Ho = 2 * H, Wo = 2 * Wd;
  const int HoWo = Ho * Wo;
  const int m0 = blockIdx.x * 16;
  const int n0 = (blockIdx.y * 4 + wave) * 16;
  if (n0 >= Cout) return;

  const int bb  = m0 / HoWo;                  // uniform: whole tile in one batch
  const int rs0 = m0 - bb * HoWo;
  const int rs  = rs0 + mr;
  const int oh  = rs / Wo;
  const int ow  = rs - oh * Wo;
  const int abase = bb * Cin * H * Wd;        // fits in i32

  const int Ktot = Cin * 16;
  const int oc = n0 + nc;

  f32x8 acc = {};
  for (int kb = 0; kb < Ktot; kb += 32) {
    bf16x16 af, bf;
#pragma unroll
    for (int i = 0; i < 16; ++i) {
      int ka = kb + akmap(i, half);
      int ci = ka >> 4, rr = ka & 15, kh = rr >> 2, kw = rr & 3;
      int th = oh + 1 - kh, tw = ow + 1 - kw;
      int ih = th >> 1, iw = tw >> 1;
      bool ok = (th >= 0) & (tw >= 0) & !(th & 1) & !(tw & 1) & (ih < H) & (iw < Wd);
      int aoff = abase + (ci * H + ih) * Wd + iw;
      aoff = ok ? aoff : 0;                   // v_cndmask_b32
      float av = in[aoff];
      af[i] = (__bf16)(ok ? av : 0.f);
      int kB = kb + bkmap(i, half);
      int ci2 = kB >> 4, rr2 = kB & 15;
      bf[i] = (__bf16)wt[(ci2 * Cout + oc) * 16 + rr2];
    }
    acc = wmma_bf16(af, bf, acc);
  }
  const long obase = ((long)bb * Cout + oc) * (long)HoWo + rs0 + half * 8;
  const float bsv = bias[oc];
#pragma unroll
  for (int r = 0; r < 8; ++r)
    out[obase + r] = acc[r] + bsv;
}

// ---------------------------------------------------------------------------
// Attention pieces (B=8, L=1024, C=256, 4 heads of 64).
// ---------------------------------------------------------------------------
// t: (8,256,1024) channel-major -> qkv: (8,1024,768).  wqkv: (768,256).
__global__ void __launch_bounds__(128)
qkv_gemm(const float* __restrict__ t, const float* __restrict__ wqkv,
         const float* __restrict__ bqkv, float* __restrict__ qkv)
{
  const int lane = threadIdx.x & 31, wave = threadIdx.x >> 5;
  const int half = lane >> 4, mr = lane & 15, nc = lane & 15;
  const int m0 = blockIdx.x * 16;
  const int n0 = (blockIdx.y * 4 + wave) * 16;
  const int b  = m0 >> 10;
  const int l  = (m0 & 1023) + mr;
  f32x8 acc = {};
  for (int kb = 0; kb < 256; kb += 32) {
    bf16x16 af, bf;
#pragma unroll
    for (int i = 0; i < 16; ++i) {
      af[i] = (__bf16)t[(((b << 8) + kb + akmap(i, half)) << 10) + l];
      bf[i] = (__bf16)wqkv[(n0 + nc) * 256 + kb + bkmap(i, half)];
    }
    acc = wmma_bf16(af, bf, acc);
  }
#pragma unroll
  for (int r = 0; r < 8; ++r) {
    int m = m0 + r + half * 8;
    int n = n0 + nc;
    qkv[(long)m * 768 + n] = acc[r] + bqkv[n];
  }
}

// S[b,h,q,k] = (Q.K^T)/8 over d=64
__global__ void __launch_bounds__(128)
attn_scores(const float* __restrict__ qkv, float* __restrict__ s)
{
  const int lane = threadIdx.x & 31, wave = threadIdx.x >> 5;
  const int half = lane >> 4, mr = lane & 15, nc = lane & 15;
  const int bh = blockIdx.z, b = bh >> 2, h = bh & 3;
  const int q0 = blockIdx.x * 16;
  const int k0 = (blockIdx.y * 4 + wave) * 16;
  const int qoff = h * 64, koff = 256 + h * 64;
  f32x8 acc = {};
#pragma unroll
  for (int kb = 0; kb < 64; kb += 32) {
    bf16x16 af, bf;
#pragma unroll
    for (int i = 0; i < 16; ++i) {
      af[i] = (__bf16)qkv[(long)((b << 10) + q0 + mr) * 768 + qoff + kb + akmap(i, half)];
      bf[i] = (__bf16)qkv[(long)((b << 10) + k0 + nc) * 768 + koff + kb + bkmap(i, half)];
    }
    acc = wmma_bf16(af, bf, acc);
  }
  const long base = ((long)bh << 20);
#pragma unroll
  for (int r = 0; r < 8; ++r) {
    int q = q0 + r + half * 8, k = k0 + nc;
    s[base + ((long)q << 10) + k] = acc[r] * 0.125f;
  }
}

__global__ void softmax_rows(float* __restrict__ s, int L)
{
  float* p = s + (long)blockIdx.x * L;
  __shared__ float red[256];
  float m = -1e30f;
  for (int i = threadIdx.x; i < L; i += 256) m = fmaxf(m, p[i]);
  red[threadIdx.x] = m; __syncthreads();
  for (int st = 128; st > 0; st >>= 1) {
    if (threadIdx.x < st) red[threadIdx.x] = fmaxf(red[threadIdx.x], red[threadIdx.x + st]);
    __syncthreads();
  }
  m = red[0]; __syncthreads();
  float sum = 0.f;
  for (int i = threadIdx.x; i < L; i += 256) { float e = __expf(p[i] - m); p[i] = e; sum += e; }
  red[threadIdx.x] = sum; __syncthreads();
  for (int st = 128; st > 0; st >>= 1) {
    if (threadIdx.x < st) red[threadIdx.x] += red[threadIdx.x + st];
    __syncthreads();
  }
  const float inv = 1.f / red[0];
  for (int i = threadIdx.x; i < L; i += 256) p[i] *= inv;
}

// O = P(LxL) @ V(Lx64) per (b,h); output AO (8,1024,256)
__global__ void __launch_bounds__(128)
attn_out(const float* __restrict__ s, const float* __restrict__ qkv,
         float* __restrict__ ao)
{
  const int lane = threadIdx.x & 31, wave = threadIdx.x >> 5;
  const int half = lane >> 4, mr = lane & 15, nc = lane & 15;
  const int bh = blockIdx.z, b = bh >> 2, h = bh & 3;
  const int q0 = blockIdx.x * 16;
  const int n0 = wave * 16;                 // 4 waves cover d=64
  const int voff = 512 + h * 64;
  const long sbase = ((long)bh << 20);
  f32x8 acc = {};
#pragma unroll 2
  for (int kb = 0; kb < 1024; kb += 32) {
    bf16x16 af, bf;
#pragma unroll
    for (int i = 0; i < 16; ++i) {
      af[i] = (__bf16)s[sbase + ((long)(q0 + mr) << 10) + kb + akmap(i, half)];
      bf[i] = (__bf16)qkv[(long)((b << 10) + kb + bkmap(i, half)) * 768 + voff + n0 + nc];
    }
    acc = wmma_bf16(af, bf, acc);
  }
#pragma unroll
  for (int r = 0; r < 8; ++r) {
    int q = q0 + r + half * 8;
    ao[(long)((b << 10) + q) * 256 + h * 64 + n0 + nc] = acc[r];
  }
}

// out(8,256,1024) = AO(8,1024,256) @ wo^T + bo + resid   (resid is channel-major x)
__global__ void __launch_bounds__(128)
proj_gemm(const float* __restrict__ ao, const float* __restrict__ wo,
          const float* __restrict__ bo, const float* __restrict__ resid,
          float* __restrict__ out)
{
  const int lane = threadIdx.x & 31, wave = threadIdx.x >> 5;
  const int half = lane >> 4, mr = lane & 15, nc = lane & 15;
  const int m0 = blockIdx.x * 16;
  const int n0 = (blockIdx.y * 4 + wave) * 16;
  const int b = m0 >> 10;
  f32x8 acc = {};
  for (int kb = 0; kb < 256; kb += 32) {
    bf16x16 af, bf;
#pragma unroll
    for (int i = 0; i < 16; ++i) {
      af[i] = (__bf16)ao[(long)((b << 10) + (m0 & 1023) + mr) * 256 + kb + akmap(i, half)];
      bf[i] = (__bf16)wo[(n0 + nc) * 256 + kb + bkmap(i, half)];
    }
    acc = wmma_bf16(af, bf, acc);
  }
#pragma unroll
  for (int r = 0; r < 8; ++r) {
    int l = (m0 & 1023) + r + half * 8;
    int c = n0 + nc;
    long idx = ((long)(b * 256 + c) << 10) + l;
    out[idx] = acc[r] + bo[c] + resid[idx];
  }
}

// ---------------------------------------------------------------------------
// VQ nearest-codebook search.  Codebook (8192x4 fp32, 128 KB) is staged into
// LDS with async global->LDS copies (ASYNCcnt path).  One thread per pixel.
// Forward value of the straight-through output is just E[argmin].
// loss = (1 + BETA) * mean((q - z)^2)
// ---------------------------------------------------------------------------
__global__ void vq_kernel(const float* __restrict__ z, const float* __restrict__ E,
                          float* __restrict__ zq, float* __restrict__ loss)
{
  extern __shared__ float cb[];                    // 8192*4 floats
  const unsigned lbase = (unsigned)(uintptr_t)cb;  // LDS byte offset (addr[31:0])
  for (int it = 0; it < 32; ++it) {
    int chunk = it * 256 + threadIdx.x;            // float4 chunk id, 0..8191
    unsigned laddr = lbase + (unsigned)chunk * 16u;
    unsigned long long gaddr = (unsigned long long)(uintptr_t)(E + (long)chunk * 4);
    asm volatile("global_load_async_to_lds_b128 %0, %1, off"
                 :: "v"(laddr), "v"(gaddr) : "memory");
  }
  asm volatile("s_wait_asynccnt 0" ::: "memory");
  __syncthreads();

  const int p  = blockIdx.x * 256 + threadIdx.x;   // 0..8191 pixels
  const int b  = p >> 10;
  const int sp = p & 1023;
  const float z0 = z[((b * 4 + 0) << 10) + sp];
  const float z1 = z[((b * 4 + 1) << 10) + sp];
  const float z2 = z[((b * 4 + 2) << 10) + sp];
  const float z3 = z[((b * 4 + 3) << 10) + sp];

  int best = 0; float bestd = 3.4e38f;
  for (int j = 0; j < 8192; ++j) {
    float d0 = cb[j * 4 + 0] - z0;
    float d1 = cb[j * 4 + 1] - z1;
    float d2 = cb[j * 4 + 2] - z2;
    float d3 = cb[j * 4 + 3] - z3;
    float d = d0 * d0 + d1 * d1 + d2 * d2 + d3 * d3;
    if (d < bestd) { bestd = d; best = j; }
  }
  const float q0 = cb[best * 4 + 0], q1 = cb[best * 4 + 1];
  const float q2 = cb[best * 4 + 2], q3 = cb[best * 4 + 3];
  zq[((b * 4 + 0) << 10) + sp] = q0;
  zq[((b * 4 + 1) << 10) + sp] = q1;
  zq[((b * 4 + 2) << 10) + sp] = q2;
  zq[((b * 4 + 3) << 10) + sp] = q3;
  float l = (q0 - z0) * (q0 - z0) + (q1 - z1) * (q1 - z1)
          + (q2 - z2) * (q2 - z2) + (q3 - z3) * (q3 - z3);
  atomicAdd(loss, l * (1.2f / 32768.f));           // (1+BETA) * mean over 8192*4
}

// ---------------------------------------------------------------------------
// Host orchestration.
// Input leaf order assumed = jax.tree_util (sorted dict keys at every level):
//  0 codebook | 1,2 conv_in w,b | 3,4 dec_conv_in | 5,6 dec_conv_out |
//  7,8 dec_norm g,b | 9 d1.db 10 d1.dw 11..20 d1.res(b1,b2,c1,c2,cr,g1,g2,w1,w2,wr) |
//  21,22 d2.db,dw 23..32 d2.res | 33,34 enc_conv_out | 35,36 enc_norm |
//  37..42 mid_dec.attn(bo,bqkv,nb,ng,wo,wqkv) 43..52 mid_dec.res0 53..62 mid_dec.res1 |
//  63..68 mid_enc.attn 69..78 mid_enc.res0 79..88 mid_enc.res1 |
//  89,90 post_quant | 91,92 pre_quant | 93..102 up1.res 103 up1.ub 104 up1.uw |
//  105..114 up2.res 115 up2.ub 116 up2.uw | 117 x
// ---------------------------------------------------------------------------
extern "C" void kernel_launch(void* const* d_in, const int* in_sizes, int n_in,
                              void* d_out, int out_size, void* d_ws, size_t ws_size,
                              hipStream_t stream)
{
  (void)in_sizes; (void)n_in; (void)out_size; (void)ws_size;
  auto F = [&](int i) { return (const float*)d_in[i]; };

  const size_t BIG = 33554432ull;          // 8*256*128*128 floats (134 MB)
  float* W0 = (float*)d_ws;
  float* W1 = W0 + BIG;
  float* W2 = W1 + BIG;
  float* W3 = W2 + BIG;
  float* W4 = W3 + BIG;

  auto gn = [&](const float* x, const float* g, const float* b, float* y,
                int C, int HW, bool silu) {
    gn_kernel<<<dim3(8 * 32), 256, 0, stream>>>(x, g, b, y, C, HW, silu ? 1 : 0);
  };
  auto conv3 = [&](const float* in, const float* w, const float* bias, const float* res,
                   float* out, int Cin, int H, int Cout) {
    dim3 grid((8 * H * H) / 16, (Cout + 63) / 64);
    conv2d_wmma<3, 3><<<grid, 128, 0, stream>>>(in, w, bias, res, out,
                                                8, Cin, H, H, Cout, H, H, 1, 1);
  };
  auto conv1 = [&](const float* in, const float* w, const float* bias, float* out,
                   int Cin, int H, int Cout) {
    dim3 grid((8 * H * H) / 16, (Cout + 63) / 64);
    conv2d_wmma<1, 1><<<grid, 128, 0, stream>>>(in, w, bias, nullptr, out,
                                                8, Cin, H, H, Cout, H, H, 1, 0);
  };
  auto conv4 = [&](const float* in, const float* w, const float* bias, float* out,
                   int Cin, int H, int Cout) {
    int Ho = H / 2;
    dim3 grid((8 * Ho * Ho) / 16, (Cout + 63) / 64);
    conv2d_wmma<4, 4><<<grid, 128, 0, stream>>>(in, w, bias, nullptr, out,
                                                8, Cin, H, H, Cout, Ho, Ho, 2, 1);
  };
  auto convt = [&](const float* in, const float* w, const float* bias, float* out,
                   int Cin, int H, int Cout) {
    int Ho = 2 * H;
    dim3 grid((8 * Ho * Ho) / 16, (Cout + 63) / 64);
    convt_wmma<<<grid, 128, 0, stream>>>(in, w, bias, out, 8, Cin, H, H, Cout);
  };
  // res leaves: base+0..9 = b1,b2,c1,c2,cr,g1,g2,w1,w2,wr
  auto resnet = [&](const float* xin, float* out, int Cin, int Cout, int H, int base) {
    gn(xin, F(base + 5), F(base + 0), W1, Cin, H * H, true);
    conv3(W1, F(base + 7), F(base + 2), nullptr, W2, Cin, H, Cout);
    gn(W2, F(base + 6), F(base + 1), W1, Cout, H * H, true);
    conv1(xin, F(base + 9), F(base + 4), W3, Cin, H, Cout);
    conv3(W1, F(base + 8), F(base + 3), W3, out, Cout, H, Cout);
  };
  // attn leaves: base+0..5 = bo,bqkv,nb,ng,wo,wqkv ; x at 32x32, C=256
  auto attn = [&](float* xin, float* out, int base) {
    gn(xin, F(base + 3), F(base + 2), W1, 256, 1024, false);
    qkv_gemm<<<dim3(512, 12), 128, 0, stream>>>(W1, F(base + 5), F(base + 1), W2);
    attn_scores<<<dim3(64, 16, 32), 128, 0, stream>>>(W2, W3);
    softmax_rows<<<32768, 256, 0, stream>>>(W3, 1024);
    attn_out<<<dim3(64, 1, 32), 128, 0, stream>>>(W3, W2, W1);
    proj_gemm<<<dim3(512, 4), 128, 0, stream>>>(W1, F(base + 4), F(base + 0), xin, out);
  };

  float* cur = W0;
  float* alt = W4;
  auto swp = [&]() { float* t = cur; cur = alt; alt = t; };

  const float* x = F(117);

  // ---- encoder ----
  conv3(x, F(1), F(2), nullptr, cur, 3, 128, 128);           // conv_in
  resnet(cur, cur, 128, 256, 128, 11);                        // down1.res
  conv4(cur, F(10), F(9), alt, 256, 128, 256); swp();         // down1 downsample -> 64
  resnet(cur, cur, 256, 256, 64, 23);                         // down2.res
  conv4(cur, F(22), F(21), alt, 256, 64, 256); swp();         // down2 downsample -> 32
  resnet(cur, cur, 256, 256, 32, 69);                         // mid_enc.res0
  attn(cur, alt, 63); swp();                                  // mid_enc.attn
  resnet(cur, cur, 256, 256, 32, 79);                         // mid_enc.res1
  gn(cur, F(35), F(36), W1, 256, 1024, true);                 // enc_norm + silu
  conv3(W1, F(33), F(34), nullptr, W2, 256, 32, 4);           // enc_conv_out -> (8,4,32,32)
  conv1(W2, F(91), F(92), W1, 4, 32, 4);                      // pre_quant

  // ---- quantize ----
  float* loss_ptr = (float*)d_out + 393216;                   // 8*3*128*128
  hipMemsetAsync(loss_ptr, 0, sizeof(float), stream);
  vq_kernel<<<32, 256, 8192 * 4 * sizeof(float), stream>>>(W1, F(0), W2, loss_ptr);

  // ---- decoder ----
  conv1(W2, F(89), F(90), W1, 4, 32, 4);                      // post_quant
  conv3(W1, F(3), F(4), nullptr, cur, 4, 32, 256);            // dec_conv_in
  resnet(cur, cur, 256, 256, 32, 43);                         // mid_dec.res0
  attn(cur, alt, 37); swp();                                  // mid_dec.attn
  resnet(cur, cur, 256, 256, 32, 53);                         // mid_dec.res1
  convt(cur, F(104), F(103), alt, 256, 32, 256); swp();       // up1 upsample -> 64
  resnet(cur, cur, 256, 256, 64, 93);                         // up1.res
  convt(cur, F(116), F(115), alt, 256, 64, 256); swp();       // up2 upsample -> 128
  resnet(cur, cur, 256, 128, 128, 105);                       // up2.res (256->128)
  gn(cur, F(7), F(8), alt, 128, 16384, true);                 // dec_norm + silu
  conv3(alt, F(5), F(6), nullptr, (float*)d_out, 128, 128, 3);// dec_conv_out -> output
}
